// Attention_24893630448050
// MI455X (gfx1250) — compile-verified, matrix operand order checked by reading, not run on hardware
//
#include <hip/hip_runtime.h>

#define BATCH  2
#define SEQ    2048
#define DMODEL 2048
#define NQH    32
#define NKVH   8
#define HDIM   64
#define NREP   (NQH / NKVH)
#define NEG_INF_F (-3.0e38f)

typedef __attribute__((ext_vector_type(16))) __bf16 v16bf;
typedef __attribute__((ext_vector_type(8)))  float  v8f;
typedef __attribute__((ext_vector_type(4)))  unsigned int u32x4;

// Load 16 bf16 elements from two 16-byte-aligned chunks.
static __device__ __forceinline__ v16bf load_bf16_pair(const __bf16* p0, const __bf16* p1) {
  union { u32x4 u[2]; v16bf v; } r;
  r.u[0] = *(const u32x4*)p0;
  r.u[1] = *(const u32x4*)p1;
  return r.v;
}
static __device__ __forceinline__ v16bf load_bf16x16(const __bf16* p) {
  return load_bf16_pair(p, p + 8);
}
static __device__ __forceinline__ v8f wmma_bf16(v16bf a, v16bf b, v8f c) {
  return __builtin_amdgcn_wmma_f32_16x16x32_bf16(false, a, false, b, (short)0, c, false, false);
}

// ---------------------------------------------------------------------------
// f32 -> bf16 conversion
// ---------------------------------------------------------------------------
__global__ void __launch_bounds__(256) cvt_f32_bf16(const float* __restrict__ src,
                                                    __bf16* __restrict__ dst, int n) {
  int i = blockIdx.x * 256 + threadIdx.x;
  if (i < n) dst[i] = (__bf16)src[i];
}

// ---------------------------------------------------------------------------
// RoPE + layout change: src [B*S, nheads*HD] f32 -> dst [B][nheads][S][HD] bf16
// One thread per rotation pair. Q gets scale = 1/sqrt(HD).
// ---------------------------------------------------------------------------
__global__ void __launch_bounds__(256) rope_cvt(const float* __restrict__ src,
                                                const float* __restrict__ fcos,
                                                const float* __restrict__ fsin,
                                                __bf16* __restrict__ dst,
                                                int nheads, float scale) {
  size_t i = (size_t)blockIdx.x * 256 + threadIdx.x;
  size_t total = (size_t)BATCH * SEQ * nheads * (HDIM / 2);
  if (i >= total) return;
  int p  = (int)(i % (HDIM / 2)); size_t t = i / (HDIM / 2);
  int hh = (int)(t % nheads);     t /= nheads;
  int s  = (int)(t % SEQ);
  int b  = (int)(t / SEQ);
  const float* row = src + ((size_t)b * SEQ + s) * ((size_t)nheads * HDIM) + (size_t)hh * HDIM;
  float xr = row[2 * p], xi = row[2 * p + 1];
  float c  = fcos[(size_t)s * (HDIM / 2) + p];
  float sn = fsin[(size_t)s * (HDIM / 2) + p];
  float orr = (xr * c - xi * sn) * scale;
  float oii = (xr * sn + xi * c) * scale;
  __bf16* drow = dst + (((size_t)b * nheads + hh) * SEQ + s) * HDIM;
  drow[2 * p]     = (__bf16)orr;
  drow[2 * p + 1] = (__bf16)oii;
}

// ---------------------------------------------------------------------------
// V transpose: src [B*S, KVH*HD] f32 -> dst [B][KVH][HD][S] bf16
// ---------------------------------------------------------------------------
__global__ void __launch_bounds__(256) v_transpose(const float* __restrict__ src,
                                                   __bf16* __restrict__ dst) {
  size_t i = (size_t)blockIdx.x * 256 + threadIdx.x;
  size_t total = (size_t)BATCH * SEQ * NKVH * HDIM;
  if (i >= total) return;
  int d  = (int)(i % HDIM); size_t t = i / HDIM;
  int hh = (int)(t % NKVH); t /= NKVH;
  int s  = (int)(t % SEQ);
  int b  = (int)(t / SEQ);
  float v = src[((size_t)b * SEQ + s) * (NKVH * HDIM) + (size_t)hh * HDIM + d];
  dst[(((size_t)b * NKVH + hh) * HDIM + d) * SEQ + s] = (__bf16)v;
}

// ---------------------------------------------------------------------------
// Generic NT GEMM: C[M,N](f32) = A[M,K](bf16) * B[N,K](bf16)^T
// Block = 256 threads = 8 waves (4 along M x 2 along N), block tile 128x128.
// Wave tile 32x64. K must be a multiple of 32; M,N multiples of 128.
// Both A and B fragments use the same contiguous k-permutation, which is
// legal since the reduction index mapping only has to agree between A and B.
// ---------------------------------------------------------------------------
__global__ void __launch_bounds__(256) gemm_nt_bf16f32(const __bf16* __restrict__ A,
                                                       const __bf16* __restrict__ B,
                                                       float* __restrict__ C,
                                                       int M, int N, int K) {
  const int lane = threadIdx.x & 31;
  const int wave = threadIdx.x >> 5;   // 0..7
  const int wm   = wave >> 1;          // 0..3
  const int wn   = wave & 1;           // 0..1
  const int m0 = blockIdx.y * 128 + wm * 32;
  const int n0 = blockIdx.x * 128 + wn * 64;
  const int lr   = lane & 15;
  const int koff = (lane >> 4) * 16;

  v8f acc[2][4];
  for (int i = 0; i < 2; ++i)
    for (int j = 0; j < 4; ++j)
      acc[i][j] = (v8f){0.f, 0.f, 0.f, 0.f, 0.f, 0.f, 0.f, 0.f};

  const __bf16* a0p = A + (size_t)(m0 + lr) * K + koff;
  const __bf16* a1p = A + (size_t)(m0 + 16 + lr) * K + koff;
  const __bf16* b0p = B + (size_t)(n0 + lr) * K + koff;
  const __bf16* b1p = B + (size_t)(n0 + 16 + lr) * K + koff;
  const __bf16* b2p = B + (size_t)(n0 + 32 + lr) * K + koff;
  const __bf16* b3p = B + (size_t)(n0 + 48 + lr) * K + koff;

  for (int k0 = 0; k0 < K; k0 += 32) {
    v16bf a0 = load_bf16x16(a0p + k0);
    v16bf a1 = load_bf16x16(a1p + k0);
    v16bf b0 = load_bf16x16(b0p + k0);
    v16bf b1 = load_bf16x16(b1p + k0);
    v16bf b2 = load_bf16x16(b2p + k0);
    v16bf b3 = load_bf16x16(b3p + k0);
    __builtin_prefetch(a0p + k0 + 128, 0, 3);   // global_prefetch_b8
    __builtin_prefetch(b0p + k0 + 128, 0, 3);
    acc[0][0] = wmma_bf16(a0, b0, acc[0][0]);
    acc[0][1] = wmma_bf16(a0, b1, acc[0][1]);
    acc[0][2] = wmma_bf16(a0, b2, acc[0][2]);
    acc[0][3] = wmma_bf16(a0, b3, acc[0][3]);
    acc[1][0] = wmma_bf16(a1, b0, acc[1][0]);
    acc[1][1] = wmma_bf16(a1, b1, acc[1][1]);
    acc[1][2] = wmma_bf16(a1, b2, acc[1][2]);
    acc[1][3] = wmma_bf16(a1, b3, acc[1][3]);
  }

  // C/D layout: lane half h = lane>>4, col n = lane&15, vgpr r -> row 8h+r.
  const int orow = (lane >> 4) * 8;
  for (int i = 0; i < 2; ++i)
    for (int j = 0; j < 4; ++j)
      for (int r = 0; r < 8; ++r)
        C[(size_t)(m0 + i * 16 + orow + r) * N + n0 + j * 16 + lr] = acc[i][j][r];
}

// ---------------------------------------------------------------------------
// Flash attention (causal + pad mask). One wave owns 16 queries.
// Q pre-scaled by 1/sqrt(HD). Scores computed transposed: St = K * Q^T, so
// each lane holds a ROW of P (fixed query) -> P C-fragments repack into
// A-fragments of the P*V GEMM without cross-lane moves.
// Qb [B][H][S][HD], Kb [B][KVH][S][HD], Vt [B][KVH][HD][S], out [B][S][H*HD]
// ---------------------------------------------------------------------------
__global__ void __launch_bounds__(256) attn_fwd(const __bf16* __restrict__ Qb,
                                                const __bf16* __restrict__ Kb,
                                                const __bf16* __restrict__ Vt,
                                                const int* __restrict__ attn_mask,
                                                __bf16* __restrict__ Ob) {
  const int lane = threadIdx.x & 31;
  const int wave = threadIdx.x >> 5;        // 0..7
  const int h    = blockIdx.y;
  const int b    = blockIdx.z;
  const int q0   = blockIdx.x * 128 + wave * 16;   // 16 queries per wave
  const int kvh  = h / NREP;
  const int lr   = lane & 15;
  const int lh   = lane >> 4;

  const __bf16* Qp = Qb + (((size_t)b * NQH + h) * SEQ + q0) * HDIM;
  const __bf16* Kp = Kb + (((size_t)b * NKVH + kvh) * SEQ) * HDIM;
  const __bf16* Vp = Vt + (((size_t)b * NKVH + kvh) * HDIM) * SEQ;
  const int* msk = attn_mask + (size_t)b * SEQ;

  // Q B-fragments (reused every key step): lane -> Q row (q0+lr), d chunk.
  v16bf qf[2];
  for (int c = 0; c < 2; ++c)
    qf[c] = load_bf16x16(Qp + (size_t)lr * HDIM + c * 32 + lh * 16);

  v8f o[4];
  for (int g = 0; g < 4; ++g)
    o[g] = (v8f){0.f, 0.f, 0.f, 0.f, 0.f, 0.f, 0.f, 0.f};
  float m_i = NEG_INF_F;
  float l_i = 0.f;

  const int qidx = q0 + lr;     // this lane's query (St column)
  const int kend = q0 + 16;     // causal: need keys 0 .. q0+15

  for (int k0 = 0; k0 < kend; k0 += 64) {
    // ---- St = K * Q^T for 64 keys x 16 queries -------------------------
    v8f st[4];
    for (int f = 0; f < 4; ++f)
      st[f] = (v8f){0.f, 0.f, 0.f, 0.f, 0.f, 0.f, 0.f, 0.f};
    for (int c = 0; c < 2; ++c) {
      for (int f = 0; f < 4; ++f) {
        v16bf kf = load_bf16x16(Kp + (size_t)(k0 + f * 16 + lr) * HDIM + c * 32 + lh * 16);
        st[f] = wmma_bf16(kf, qf[c], st[f]);
      }
    }
    // ---- mask + running row max ---------------------------------------
    float mx = m_i;
    for (int f = 0; f < 4; ++f) {
      for (int r = 0; r < 8; ++r) {
        int key = k0 + f * 16 + lh * 8 + r;
        float s = st[f][r];
        if (key > qidx || msk[key] == 0) s = NEG_INF_F;
        st[f][r] = s;
        mx = fmaxf(mx, s);
      }
    }
    mx = fmaxf(mx, __shfl_xor(mx, 16, 32));   // lanes l and l^16 hold the other keys
    float corr = __expf(m_i - mx);
    m_i = mx;
    // ---- exponentiate, row sum ----------------------------------------
    float lsum = 0.f;
    for (int f = 0; f < 4; ++f) {
      for (int r = 0; r < 8; ++r) {
        float p = __expf(st[f][r] - mx);
        st[f][r] = p;
        lsum += p;
      }
    }
    lsum += __shfl_xor(lsum, 16, 32);
    l_i = l_i * corr + lsum;
    // ---- pack P into A-fragments (pure in-lane repack) ----------------
    // k-map for PV gemm: e<8 -> 8h+e ; e>=8 -> 16+8h+(e-8)
    v16bf pa[2];
    for (int e = 0; e < 8; ++e) {
      pa[0][e]     = (__bf16)st[0][e];
      pa[0][8 + e] = (__bf16)st[1][e];
      pa[1][e]     = (__bf16)st[2][e];
      pa[1][8 + e] = (__bf16)st[3][e];
    }
    // ---- rescale O rows by corr (per-query broadcast via shuffle) -----
    float cr[8];
    for (int r = 0; r < 8; ++r)
      cr[r] = __shfl(corr, lh * 8 + r, 32);
    for (int g = 0; g < 4; ++g)
      for (int r = 0; r < 8; ++r)
        o[g][r] *= cr[r];
    // ---- O += P * V  (k-dim 64 = 2 steps of 32) -----------------------
    for (int ks = 0; ks < 2; ++ks) {
      const int kb = k0 + ks * 32;
      for (int g = 0; g < 4; ++g) {
        // B-frag: lane holds Vt row (16g+lr), cols kb+8h.. and kb+16+8h..
        const __bf16* vp = Vp + (size_t)(g * 16 + lr) * SEQ + kb + lh * 8;
        v16bf vb = load_bf16_pair(vp, vp + 16);
        o[g] = wmma_bf16(pa[ks], vb, o[g]);
      }
    }
  }

  // ---- epilogue: divide by row sum, store bf16 [B][S][H*HD] -----------
  float lrow[8];
  for (int r = 0; r < 8; ++r)
    lrow[r] = __shfl(l_i, lh * 8 + r, 32);
  for (int g = 0; g < 4; ++g) {
    for (int r = 0; r < 8; ++r) {
      float val = o[g][r] / lrow[r];
      size_t row = (size_t)b * SEQ + q0 + lh * 8 + r;
      Ob[row * (NQH * HDIM) + (size_t)h * HDIM + g * 16 + lr] = (__bf16)val;
    }
  }
}

// ---------------------------------------------------------------------------
// Launcher
// ---------------------------------------------------------------------------
extern "C" void kernel_launch(void* const* d_in, const int* in_sizes, int n_in,
                              void* d_out, int out_size, void* d_ws, size_t ws_size,
                              hipStream_t stream) {
  (void)in_sizes; (void)n_in; (void)out_size; (void)ws_size;
  const float* x    = (const float*)d_in[0];
  const float* fcos = (const float*)d_in[1];
  const float* fsin = (const float*)d_in[2];
  const float* wq   = (const float*)d_in[3];
  const float* wk   = (const float*)d_in[4];
  const float* wv   = (const float*)d_in[5];
  const float* wo   = (const float*)d_in[6];
  const int*   am   = (const int*)d_in[7];
  float* out = (float*)d_out;

  char* ws = (char*)d_ws;
  size_t off = 0;
  auto alloc = [&](size_t bytes) { char* p = ws + off; off += (bytes + 255) & ~(size_t)255; return p; };

  const size_t MS = (size_t)BATCH * SEQ;             // 4096 rows
  __bf16* xb   = (__bf16*)alloc(MS * DMODEL * 2);
  __bf16* wqb  = (__bf16*)alloc((size_t)DMODEL * DMODEL * 2);
  __bf16* wkb  = (__bf16*)alloc((size_t)NKVH * HDIM * DMODEL * 2);
  __bf16* wvb  = (__bf16*)alloc((size_t)NKVH * HDIM * DMODEL * 2);
  __bf16* wob  = (__bf16*)alloc((size_t)DMODEL * DMODEL * 2);
  float*  qf32 = (float*)alloc(MS * DMODEL * 4);
  float*  kf32 = (float*)alloc(MS * NKVH * HDIM * 4);
  float*  vf32 = (float*)alloc(MS * NKVH * HDIM * 4);
  __bf16* qbf  = (__bf16*)alloc(MS * DMODEL * 2);
  __bf16* kbf  = (__bf16*)alloc(MS * NKVH * HDIM * 2);
  __bf16* vtb  = (__bf16*)alloc(MS * NKVH * HDIM * 2);
  __bf16* aob  = (__bf16*)alloc(MS * DMODEL * 2);

  auto cvt = [&](const float* s, __bf16* d, int n) {
    cvt_f32_bf16<<<(n + 255) / 256, 256, 0, stream>>>(s, d, n);
  };
  cvt(x,  xb,  (int)(MS * DMODEL));
  cvt(wq, wqb, DMODEL * DMODEL);
  cvt(wk, wkb, NKVH * HDIM * DMODEL);
  cvt(wv, wvb, NKVH * HDIM * DMODEL);
  cvt(wo, wob, DMODEL * DMODEL);

  // QKV projections (NT GEMMs)
  gemm_nt_bf16f32<<<dim3(DMODEL / 128, (int)(MS / 128)), 256, 0, stream>>>(
      xb, wqb, qf32, (int)MS, DMODEL, DMODEL);
  gemm_nt_bf16f32<<<dim3((NKVH * HDIM) / 128, (int)(MS / 128)), 256, 0, stream>>>(
      xb, wkb, kf32, (int)MS, NKVH * HDIM, DMODEL);
  gemm_nt_bf16f32<<<dim3((NKVH * HDIM) / 128, (int)(MS / 128)), 256, 0, stream>>>(
      xb, wvb, vf32, (int)MS, NKVH * HDIM, DMODEL);

  // RoPE (+1/sqrt(HD) folded into Q) and V transpose
  {
    size_t tq = (size_t)BATCH * SEQ * NQH * (HDIM / 2);
    rope_cvt<<<(unsigned)((tq + 255) / 256), 256, 0, stream>>>(qf32, fcos, fsin, qbf, NQH, 0.125f);
    size_t tk = (size_t)BATCH * SEQ * NKVH * (HDIM / 2);
    rope_cvt<<<(unsigned)((tk + 255) / 256), 256, 0, stream>>>(kf32, fcos, fsin, kbf, NKVH, 1.0f);
    size_t tv = (size_t)BATCH * SEQ * NKVH * HDIM;
    v_transpose<<<(unsigned)((tv + 255) / 256), 256, 0, stream>>>(vf32, vtb);
  }

  // Flash attention
  attn_fwd<<<dim3(SEQ / 128, NQH, BATCH), 256, 0, stream>>>(qbf, kbf, vtb, am, aob);

  // Output projection -> f32 result
  gemm_nt_bf16f32<<<dim3(DMODEL / 128, (int)(MS / 128)), 256, 0, stream>>>(
      aob, wob, out, (int)MS, DMODEL, DMODEL);
}
